// ODEModel_44624710205573
// MI455X (gfx1250) — compile-verified, matrix operand order checked by reading, not run on hardware
//
#include <hip/hip_runtime.h>
#include <hip/hip_bf16.h>

// ---------------------------------------------------------------------------
// Neural-ODE (RK4, 127 steps) + fused output projection + collapsed LSTM head
// for MI455X (gfx1250, wave32, WMMA 16x16x32 f16->f32, TDM weight staging).
//
// Sizes: B=1024 T=128 LAT=128 HODE=128 HID=256 OUT=128 H=64 NC=2
// d_out = [ xs : 128*1024*128 f32 | lstm_out : 1024*2 f32 ]
// d_ws  = [ W1 frags 32KB | W2 frags 32KB | L2H frags 64KB | H2O frags 64KB |
//           y_final f32 512KB ]  (≈ 704 KB total)
//
// Each one-wave block DMAs the 64KB W1+W2 fragment blob into LDS once via the
// Tensor Data Mover (tensor_load_to_lds + s_wait_tensorcnt), then runs all
// 127 RK4 steps with WMMA B-fragments fed from LDS (ds_load_b128).
// ---------------------------------------------------------------------------

typedef __attribute__((ext_vector_type(16))) _Float16 v16h;
typedef __attribute__((ext_vector_type(8)))  float    v8f;
typedef unsigned int u32x4 __attribute__((ext_vector_type(4)));
typedef int          i32x8 __attribute__((ext_vector_type(8)));
typedef int          i32x4 __attribute__((ext_vector_type(4)));

#define XS_ELEMS (128 * 1024 * 128)
#define WS_W1    0
#define WS_W2    32768
#define WS_L2H   65536
#define WS_H2O   131072
#define WS_YFIN  196608

// Dynamic LDS partition (per block): [0,64K) W1+W2 frags, then staging.
#define LDS_W       0
#define LDS_YIN     65536
#define LDS_H       69632
#define LDS_TOTAL   77824

__device__ __forceinline__ v8f vzero8() {
  v8f z;
#pragma unroll
  for (int i = 0; i < 8; ++i) z[i] = 0.0f;
  return z;
}

// ---------------------------------------------------------------------------
// Pack a row-major f32 weight matrix [R,C] into fragment-major f16 tiles.
// Fragment (nt,kc) = rows [nt*16, nt*16+16), cols [kc*32, kc*32+32).
// Within a fragment: lane l holds 16 contiguous halves (row l&15,
// col-half (l>>4)), so a wave reads 1 KB fully coalesced per fragment.
// ---------------------------------------------------------------------------
__global__ void pack_w_f16_kernel(const float* __restrict__ src,
                                  _Float16* __restrict__ dst, int R, int C) {
  int idx = blockIdx.x * blockDim.x + threadIdx.x;
  if (idx >= R * C) return;
  int frag = idx >> 9;        // 512 halves per fragment
  int w    = idx & 511;
  int l    = w >> 4;
  int e    = w & 15;
  int KC   = C >> 5;
  int nt   = frag / KC;
  int kc   = frag - nt * KC;
  int row  = nt * 16 + (l & 15);
  int col  = kc * 32 + ((l >> 4) << 4) + e;
  dst[idx] = (_Float16)src[row * C + col];
}

// ---------------------------------------------------------------------------
// One-wave GEMM: D[16,NT*16] = A[16,K] (f16, LDS) x W^T (f16 fragments,
// LDS or global -- address space inferred after inlining).
//   MODE 0: +bias, ELU, store f16 to LDS (stride OUTC)
//   MODE 1: +bias,      store f16 to LDS (stride OUTC)
//   MODE 2: +bias,      store f32 to global (row stride 128)
//   MODE 3: +bias,      keep in regOut (D-layout v8f tiles)
// ---------------------------------------------------------------------------
template <int K, int NT, int MODE, int OUTC>
__device__ __forceinline__ void wave_gemm(const _Float16* __restrict__ Ain,
                                          const _Float16* __restrict__ wfrag,
                                          const float* __restrict__ bias,
                                          _Float16* __restrict__ ldsOut,
                                          float* __restrict__ gOut,
                                          v8f* __restrict__ regOut) {
  constexpr int KC = K / 32;
  const int lane = (int)(threadIdx.x & 31u);
  const int c0 = lane & 15;
  const int mb = (lane >> 4) << 3;

  v16h a[KC];
#pragma unroll
  for (int kc = 0; kc < KC; ++kc)
    a[kc] = *(const v16h*)(Ain + c0 * K + kc * 32 + (mb << 1));

#pragma unroll
  for (int nt = 0; nt < NT; ++nt) {
    v8f acc = vzero8();
#pragma unroll
    for (int kc = 0; kc < KC; ++kc) {
      v16h b = *(const v16h*)(wfrag + (size_t)(nt * KC + kc) * 512u +
                              (size_t)lane * 16u);
      acc = __builtin_amdgcn_wmma_f32_16x16x32_f16(false, a[kc], false, b,
                                                   (short)0, acc, false, false);
    }
    const float bv = bias[nt * 16 + c0];
#pragma unroll
    for (int r = 0; r < 8; ++r) {
      float v = acc[r] + bv;
      if constexpr (MODE == 0) {
        v = v > 0.0f ? v : (__expf(v) - 1.0f);  // ELU
        ldsOut[(mb + r) * OUTC + nt * 16 + c0] = (_Float16)v;
      } else if constexpr (MODE == 1) {
        ldsOut[(mb + r) * OUTC + nt * 16 + c0] = (_Float16)v;
      } else if constexpr (MODE == 2) {
        gOut[(mb + r) * 128 + nt * 16 + c0] = v;
      } else {
        acc[r] = v;
      }
    }
    if constexpr (MODE == 3) regOut[nt] = acc;
  }
}

// Stage D-layout f32 register tiles (y + c*k) as f16 row-major [16,128] in LDS.
__device__ __forceinline__ void stage_f16(_Float16* __restrict__ dst,
                                          const v8f* __restrict__ y,
                                          const v8f* __restrict__ k, float c) {
  const int lane = (int)(threadIdx.x & 31u);
  const int c0 = lane & 15;
  const int mb = (lane >> 4) << 3;
#pragma unroll
  for (int nt = 0; nt < 8; ++nt)
#pragma unroll
    for (int r = 0; r < 8; ++r) {
      float v = y[nt][r];
      if (k) v += c * k[nt][r];
      dst[(mb + r) * 128 + nt * 16 + c0] = (_Float16)v;
    }
}

// Issue one TDM descriptor: copy `nbytes` (contiguous, multiple of 32KB here)
// from global `gsrc` into LDS byte offset `ldsoff`, as a 16384-wide 2D tile of
// 2-byte elements (rows = nbytes/32768). See CDNA5 ISA.
__device__ __forceinline__ void tdm_load_to_lds(unsigned ldsoff,
                                                const void* gsrc,
                                                unsigned rows) {
  unsigned long long ga = (unsigned long long)(size_t)gsrc;
  u32x4 g0;
  g0[0] = 1u;                                     // count=1, user descriptor
  g0[1] = ldsoff;                                 // lds_addr (bytes)
  g0[2] = (unsigned)(ga & 0xffffffffu);           // global_addr[31:0]
  g0[3] = (unsigned)((ga >> 32) & 0x1ffffffu) | (2u << 30);  // addr hi | type=2
  i32x8 g1;
  g1[0] = 0x00010000;          // workgroup_mask=0, data_size=1 (2 bytes)
  g1[1] = (int)0x40000000;     // tensor_dim0[15:0]=16384 in bits[63:48]
  g1[2] = (int)(rows << 16);   // tensor_dim0 hi=0 | tensor_dim1 lo=rows
  g1[3] = (int)0x40000000;     // tensor_dim1 hi=0 | tile_dim0=16384
  g1[4] = (int)rows;           // tile_dim1=rows, tile_dim2=0
  g1[5] = 16384;               // tensor_dim0_stride[31:0]
  g1[6] = 0;                   // stride0 hi | stride1 lo
  g1[7] = 0;
  i32x4 gz;
  gz[0] = 0; gz[1] = 0; gz[2] = 0; gz[3] = 0;     // groups 2/3: unused (2D)
#if __clang_major__ >= 23
  i32x8 gz8;
#pragma unroll
  for (int i = 0; i < 8; ++i) gz8[i] = 0;
  __builtin_amdgcn_tensor_load_to_lds(g0, g1, gz, gz, gz8, 0);
#else
  __builtin_amdgcn_tensor_load_to_lds(g0, g1, gz, gz, 0);
#endif
}

// ---------------------------------------------------------------------------
// Fused RK4 integration + output projection. One wave (32 threads) per block,
// 16 batch rows per wave, 64 blocks. State lives in VGPRs for all 127 steps;
// ODE weights live in LDS (TDM-staged); projection weights stream from L2.
// ---------------------------------------------------------------------------
__global__ void __launch_bounds__(32) ode_rk4_fused_kernel(
    const float* __restrict__ y0g, const float* __restrict__ ts,
    const float* __restrict__ b1, const float* __restrict__ b2,
    const float* __restrict__ l2hb, const float* __restrict__ h2ob,
    char* __restrict__ ws, float* __restrict__ out) {
  extern __shared__ __attribute__((aligned(32))) char smem[];
  _Float16* wlds = (_Float16*)(smem + LDS_W);     // W1 frags, then W2 frags
  _Float16* sYin = (_Float16*)(smem + LDS_YIN);   // 4 KB staging
  _Float16* sH   = (_Float16*)(smem + LDS_H);     // 8 KB staging

  const int lane = (int)(threadIdx.x & 31u);
  const int m0 = blockIdx.x * 16;
  const int c0 = lane & 15;
  const int mb = (lane >> 4) << 3;

  const _Float16* w1f  = wlds;                       // 32 frags
  const _Float16* w2f  = wlds + 16384;               // 32 frags
  const _Float16* l2hf = (const _Float16*)(ws + WS_L2H);
  const _Float16* h2of = (const _Float16*)(ws + WS_H2O);
  float* yfin = (float*)(ws + WS_YFIN);

  // DMA the 64KB W1+W2 fragment blob into LDS (one TDM op per block).
  tdm_load_to_lds((unsigned)(size_t)smem + LDS_W, ws + WS_W1, 4u);

  v8f y[8], ya[8], kk[8];

  // Load initial state into WMMA D-layout tiles (overlaps with TDM).
#pragma unroll
  for (int nt = 0; nt < 8; ++nt)
#pragma unroll
    for (int r = 0; r < 8; ++r)
      y[nt][r] = y0g[(size_t)(m0 + mb + r) * 128 + nt * 16 + c0];

  __builtin_amdgcn_s_wait_tensorcnt(0);  // weights resident in LDS

  auto odef = [&](v8f* kout) {
    // h = ELU(yin @ W1^T + b1) ; kout = h @ W2^T + b2
    wave_gemm<128, 8, 0, 128>(sYin, w1f, b1, sH, nullptr, nullptr);
    wave_gemm<128, 8, 3, 0>(sH, w2f, b2, nullptr, nullptr, kout);
  };

  auto project = [&](const v8f* yv, int t) {
    // xs[t] = (y @ l2h^T + b) @ h2o^T + b
    stage_f16(sYin, yv, nullptr, 0.0f);
    wave_gemm<128, 16, 1, 256>(sYin, l2hf, l2hb, sH, nullptr, nullptr);
    wave_gemm<256, 8, 2, 0>(sH, h2of, h2ob, nullptr,
                            out + (size_t)t * 1024 * 128 + (size_t)m0 * 128,
                            nullptr);
  };

  project(y, 0);  // zs[0] = initial_state

  for (int t = 0; t < 127; ++t) {
    const float dt = ts[t + 1] - ts[t];

    stage_f16(sYin, y, nullptr, 0.0f);
    odef(kk);  // k1
#pragma unroll
    for (int nt = 0; nt < 8; ++nt)
#pragma unroll
      for (int r = 0; r < 8; ++r) ya[nt][r] = y[nt][r] + (dt / 6.0f) * kk[nt][r];

    stage_f16(sYin, y, kk, 0.5f * dt);
    odef(kk);  // k2
#pragma unroll
    for (int nt = 0; nt < 8; ++nt)
#pragma unroll
      for (int r = 0; r < 8; ++r) ya[nt][r] += (dt / 3.0f) * kk[nt][r];

    stage_f16(sYin, y, kk, 0.5f * dt);
    odef(kk);  // k3
#pragma unroll
    for (int nt = 0; nt < 8; ++nt)
#pragma unroll
      for (int r = 0; r < 8; ++r) ya[nt][r] += (dt / 3.0f) * kk[nt][r];

    stage_f16(sYin, y, kk, dt);
    odef(kk);  // k4
#pragma unroll
    for (int nt = 0; nt < 8; ++nt)
#pragma unroll
      for (int r = 0; r < 8; ++r) y[nt][r] = ya[nt][r] + (dt / 6.0f) * kk[nt][r];

    project(y, t + 1);  // zs[t+1]
  }

  // Final states feed the LSTM head.
#pragma unroll
  for (int nt = 0; nt < 8; ++nt)
#pragma unroll
    for (int r = 0; r < 8; ++r)
      yfin[(size_t)(m0 + mb + r) * 128 + nt * 16 + c0] = y[nt][r];
}

// ---------------------------------------------------------------------------
// LSTM head: the reference scans dim0=B with T+1 as batch and keeps only
// batch-row 128, whose input at step b is the final ODE state of element b.
// -> a single 1024-step recurrence over 256 gates. One 256-thread block;
// thread j owns gate j with its weight rows resident in registers (f32).
// ---------------------------------------------------------------------------
__global__ void __launch_bounds__(256) lstm_head_kernel(
    const float* __restrict__ yfin, const float* __restrict__ wih,
    const float* __restrict__ whh, const float* __restrict__ bih,
    const float* __restrict__ bhh, const float* __restrict__ fc2w,
    const float* __restrict__ fc2b, float* __restrict__ out) {
  const int j = (int)threadIdx.x;  // gate index 0..255
  __shared__ float sx[128];
  __shared__ float sh[64];
  __shared__ float sg[256];

  float wi[128];
#pragma unroll
  for (int k = 0; k < 128; ++k) wi[k] = wih[j * 128 + k];
  float wh[64];
#pragma unroll
  for (int k = 0; k < 64; ++k) wh[k] = whh[j * 64 + k];
  const float bsum = bih[j] + bhh[j];

  float c = 0.0f;  // cell state for unit j (threads 0..63)
  if (j < 64) sh[j] = 0.0f;
  __syncthreads();

  for (int b = 0; b < 1024; ++b) {
    if (j < 128) sx[j] = yfin[(size_t)b * 128 + j];
    __syncthreads();

    float acc = bsum;
#pragma unroll
    for (int k = 0; k < 128; ++k) acc += wi[k] * sx[k];
#pragma unroll
    for (int k = 0; k < 64; ++k) acc += wh[k] * sh[k];
    sg[j] = acc;
    __syncthreads();

    if (j < 64) {
      const float ig = 1.0f / (1.0f + __expf(-sg[j]));
      const float fg = 1.0f / (1.0f + __expf(-sg[j + 64]));
      const float gg = tanhf(sg[j + 128]);
      const float og = 1.0f / (1.0f + __expf(-sg[j + 192]));
      c = fg * c + ig * gg;
      sh[j] = og * tanhf(c);
    }
    __syncthreads();

    if (j < 2) {
      float o = fc2b[j];
#pragma unroll
      for (int k = 0; k < 64; ++k) o += fc2w[j * 64 + k] * sh[k];
      out[(size_t)b * 2 + j] = o;
    }
    __syncthreads();
  }
}

// ---------------------------------------------------------------------------
extern "C" void kernel_launch(void* const* d_in, const int* in_sizes, int n_in,
                              void* d_out, int out_size, void* d_ws,
                              size_t ws_size, hipStream_t stream) {
  const float* y0   = (const float*)d_in[0];
  const float* ts   = (const float*)d_in[1];
  const float* w1   = (const float*)d_in[2];
  const float* b1   = (const float*)d_in[3];
  const float* w2   = (const float*)d_in[4];
  const float* b2   = (const float*)d_in[5];
  const float* l2h  = (const float*)d_in[6];
  const float* l2hb = (const float*)d_in[7];
  const float* h2o  = (const float*)d_in[8];
  const float* h2ob = (const float*)d_in[9];
  const float* wih  = (const float*)d_in[10];
  const float* whh  = (const float*)d_in[11];
  const float* bih  = (const float*)d_in[12];
  const float* bhh  = (const float*)d_in[13];
  const float* fc2w = (const float*)d_in[14];
  const float* fc2b = (const float*)d_in[15];

  float* out = (float*)d_out;
  char* ws = (char*)d_ws;

  // Pack all four weight matrices into f16 fragment-major layout (192 KB).
  pack_w_f16_kernel<<<(128 * 128 + 255) / 256, 256, 0, stream>>>(
      w1, (_Float16*)(ws + WS_W1), 128, 128);
  pack_w_f16_kernel<<<(128 * 128 + 255) / 256, 256, 0, stream>>>(
      w2, (_Float16*)(ws + WS_W2), 128, 128);
  pack_w_f16_kernel<<<(256 * 128 + 255) / 256, 256, 0, stream>>>(
      l2h, (_Float16*)(ws + WS_L2H), 256, 128);
  pack_w_f16_kernel<<<(128 * 256 + 255) / 256, 256, 0, stream>>>(
      h2o, (_Float16*)(ws + WS_H2O), 128, 256);

  // 64 one-wave blocks, 16 batch rows each: full RK4 scan + projection.
  // 77824 B dynamic LDS: 64KB TDM-staged ODE weights + 12KB staging.
  ode_rk4_fused_kernel<<<64, 32, LDS_TOTAL, stream>>>(y0, ts, b1, b2, l2hb,
                                                      h2ob, ws, out);

  // Collapsed LSTM head -> last 2048 floats of d_out.
  lstm_head_kernel<<<1, 256, 0, stream>>>((const float*)(ws + WS_YFIN), wih,
                                          whh, bih, bhh, fc2w, fc2b,
                                          out + XS_ELEMS);
}
